// D3AT_blocks_44521630991114
// MI455X (gfx1250) — compile-verified, matrix operand order checked by reading, not run on hardware
//
#include <hip/hip_runtime.h>

// ---------------- problem constants ----------------
#define B_   2
#define L_   2048
#define C_   1024
#define H_   16
#define HD_  64
#define FF_  4096
#define M_   (B_ * L_)      // 4096 rows
#define C3_  (3 * C_)       // 3072
#define SCALE_ 0.125f       // HD^-0.5
#define EPS_ 1e-5f

typedef _Float16 v16h __attribute__((ext_vector_type(16)));
typedef _Float16 v8h  __attribute__((ext_vector_type(8)));
typedef float    v8f  __attribute__((ext_vector_type(8)));

// Load a 16-half WMMA A/B fragment: per-lane 8 contiguous halves at +half and
// 8 at +half+16 (CDNA5 ISA 16-bit A 16x32 layout; B uses the
// mirrored layout reading a [N,K] row = matrix column).
__device__ __forceinline__ v16h load_frag16(const _Float16* p, int half) {
  v8h lo = *(const v8h*)(p + half);
  v8h hi = *(const v8h*)(p + half + 16);
  return __builtin_shufflevector(lo, hi, 0,1,2,3,4,5,6,7,8,9,10,11,12,13,14,15);
}

__device__ __forceinline__ float gelu_f(float x) {
  return 0.5f * x * (1.0f + erff(x * 0.70710678118654752f));
}

// ---------------- weight convert: f32 [K,N] -> f16 transposed [N,K] ----------
__global__ void wtrans_kernel(const float* __restrict__ w, _Float16* __restrict__ wt,
                              int K, int N) {
  size_t idx = (size_t)blockIdx.x * 256 + threadIdx.x;
  if (idx >= (size_t)K * N) return;
  int n = (int)(idx / K);
  int k = (int)(idx % K);
  wt[idx] = (_Float16)w[(size_t)k * N + n];
}

// ---------------- LayerNorm over C=1024, f32 in -> f16 out -------------------
__global__ void ln_kernel(const float* __restrict__ x, const float* __restrict__ g,
                          const float* __restrict__ bt, _Float16* __restrict__ out) {
  int row = blockIdx.x, tid = threadIdx.x;
  const float4* xr = (const float4*)(x + (size_t)row * C_);
  float4 v4 = xr[tid];
  float vv[4] = {v4.x, v4.y, v4.z, v4.w};
  float s = vv[0] + vv[1] + vv[2] + vv[3];
  float ss = vv[0]*vv[0] + vv[1]*vv[1] + vv[2]*vv[2] + vv[3]*vv[3];
  __shared__ float rs[256], rq[256];
  rs[tid] = s; rq[tid] = ss;
  __syncthreads();
  for (int off = 128; off > 0; off >>= 1) {
    if (tid < off) { rs[tid] += rs[tid + off]; rq[tid] += rq[tid + off]; }
    __syncthreads();
  }
  float mean = rs[0] * (1.0f / C_);
  float var  = rq[0] * (1.0f / C_) - mean * mean;
  float rstd = rsqrtf(var + EPS_);
  int c = tid * 4;
  _Float16* orow = out + (size_t)row * C_;
  #pragma unroll
  for (int i = 0; i < 4; ++i)
    orow[c + i] = (_Float16)((vv[i] - mean) * rstd * g[c + i] + bt[c + i]);
}

// ---------------- freq bias front: rank-1 + LN + GeLU (fused) ----------------
__global__ void freqfb_kernel(const float* __restrict__ fd, const float* __restrict__ w1,
                              const float* __restrict__ b1, const float* __restrict__ g,
                              const float* __restrict__ bt, _Float16* __restrict__ out) {
  int row = blockIdx.x, tid = threadIdx.x;
  float f = fd[row];
  int c = tid * 4;
  float t[4], s = 0.f, ss = 0.f;
  #pragma unroll
  for (int i = 0; i < 4; ++i) {
    t[i] = f * w1[c + i] + b1[c + i];
    s += t[i]; ss += t[i] * t[i];
  }
  __shared__ float rs[256], rq[256];
  rs[tid] = s; rq[tid] = ss;
  __syncthreads();
  for (int off = 128; off > 0; off >>= 1) {
    if (tid < off) { rs[tid] += rs[tid + off]; rq[tid] += rq[tid + off]; }
    __syncthreads();
  }
  float mean = rs[0] * (1.0f / C_);
  float var  = rq[0] * (1.0f / C_) - mean * mean;
  float rstd = rsqrtf(var + EPS_);
  _Float16* orow = out + (size_t)row * C_;
  #pragma unroll
  for (int i = 0; i < 4; ++i) {
    float y = (t[i] - mean) * rstd * g[c + i] + bt[c + i];
    orow[c + i] = (_Float16)gelu_f(y);
  }
}

// ---------------- generic WMMA GEMM: out = act(A@Wt^T + bias [+ resid]) ------
// A: f16 [M,K] row-major (per-wave direct global loads, rows are unique per
// wave). Wt: f16 [N,K]. The 128x32 B tile is shared by all 8 waves, so it is
// staged in LDS once per workgroup via CDNA5 async global->LDS copies
// (GLOBAL_LOAD_ASYNC_TO_LDS_B128 / ASYNCcnt), double-buffered so the DMA of
// tile k+1 overlaps the 8 WMMAs of tile k. 8x cut in L2 B-traffic per block.
template<int GELU, int RESID, int OUTF16>
__global__ void gemm_kernel(const _Float16* __restrict__ A, const _Float16* __restrict__ Wt,
                            const float* __restrict__ bias, const float* __restrict__ resid,
                            void* __restrict__ outv, int Mdim, int Ndim, int Kdim) {
  __shared__ _Float16 bsm[2][128 * 32];   // double-buffered B tile [n_local][k], 2x8KB
  int tid = threadIdx.x;
  int wave = tid >> 5, lane = tid & 31;
  int col = lane & 15;
  int half  = (lane < 16) ? 0 : 8;
  int rbase = (lane < 16) ? 0 : 8;
  int m0 = blockIdx.y * 128 + wave * 16;
  int n0 = blockIdx.x * 128;
  const _Float16* abase = A + (size_t)(m0 + col) * Kdim;

  // 512 x 16B chunks per 8KB tile; 2 chunks per thread.
  int c0 = tid * 2, c1 = tid * 2 + 1;
  int nl0 = c0 >> 2, ch0 = c0 & 3;   // row (n_local), 16B chunk within 64B row
  int nl1 = c1 >> 2, ch1 = c1 & 3;

  auto issue_tile = [&](int k0, int buf) {
    const _Float16* g0 = Wt + (size_t)(n0 + nl0) * Kdim + k0 + ch0 * 8;
    const _Float16* g1 = Wt + (size_t)(n0 + nl1) * Kdim + k0 + ch1 * 8;
    unsigned l0a = (unsigned)(size_t)&bsm[buf][nl0 * 32 + ch0 * 8];
    unsigned l1a = (unsigned)(size_t)&bsm[buf][nl1 * 32 + ch1 * 8];
    asm volatile("global_load_async_to_lds_b128 %0, %1, off"
                 :: "v"(l0a), "v"((unsigned long long)(size_t)g0) : "memory");
    asm volatile("global_load_async_to_lds_b128 %0, %1, off"
                 :: "v"(l1a), "v"((unsigned long long)(size_t)g1) : "memory");
  };

  v8f acc[8] = {};
  int nk = Kdim / 32;
  issue_tile(0, 0);
  for (int kt = 0; kt < nk; ++kt) {
    int k0 = kt * 32;
    if (kt + 1 < nk) {
      issue_tile(k0 + 32, (kt + 1) & 1);
      // async loads complete in order: waiting down to 2 outstanding means the
      // 2 copies of the *current* tile have landed in LDS.
      asm volatile("s_wait_asynccnt 0x2" ::: "memory");
    } else {
      asm volatile("s_wait_asynccnt 0x0" ::: "memory");
    }
    __syncthreads();  // current tile visible to all waves
    v16h a = load_frag16(abase + k0, half);
    const _Float16* bt = &bsm[kt & 1][0];
    #pragma unroll
    for (int nt = 0; nt < 8; ++nt) {
      v16h bf = load_frag16(bt + (size_t)(nt * 16 + col) * 32, half);
      acc[nt] = __builtin_amdgcn_wmma_f32_16x16x32_f16(false, a, false, bf,
                                                       (short)0, acc[nt], false, false);
    }
    __syncthreads();  // all reads done before this buffer is overwritten (kt+2)
  }
  #pragma unroll
  for (int nt = 0; nt < 8; ++nt) {
    int n = n0 + nt * 16 + col;
    float bv = bias[n];
    #pragma unroll
    for (int r = 0; r < 8; ++r) {
      int mrow = m0 + rbase + r;
      float v = acc[nt][r] + bv;
      if (GELU)  v = gelu_f(v);
      if (RESID) v += resid[(size_t)mrow * Ndim + n];
      if (OUTF16) ((_Float16*)outv)[(size_t)mrow * Ndim + n] = (_Float16)v;
      else        ((float*)outv)[(size_t)mrow * Ndim + n] = v;
    }
  }
}

// ---------------- per-head 64x64 projection: qb/kb = fb_h @ w + b ------------
// grid: (L/64, B*H), block 128 (4 waves); out f16 [B,H,L,HD]
__global__ void headproj_kernel(const _Float16* __restrict__ fb2, const _Float16* __restrict__ wT,
                                const float* __restrict__ bias, _Float16* __restrict__ outp) {
  int bh = blockIdx.y;
  int b = bh >> 4, h = bh & 15;
  int wave = threadIdx.x >> 5, lane = threadIdx.x & 31;
  int col = lane & 15;
  int half  = (lane < 16) ? 0 : 8;
  int rbase = (lane < 16) ? 0 : 8;
  int l0 = blockIdx.x * 64 + wave * 16;
  const _Float16* abase = fb2 + (size_t)(b * L_ + l0 + col) * C_ + h * HD_;
  v8f acc[4] = {};
  #pragma unroll
  for (int kk = 0; kk < 2; ++kk) {
    v16h a = load_frag16(abase + kk * 32, half);
    #pragma unroll
    for (int nt = 0; nt < 4; ++nt) {
      v16h bf = load_frag16(wT + (size_t)(nt * 16 + col) * HD_ + kk * 32, half);
      acc[nt] = __builtin_amdgcn_wmma_f32_16x16x32_f16(false, a, false, bf,
                                                       (short)0, acc[nt], false, false);
    }
  }
  #pragma unroll
  for (int nt = 0; nt < 4; ++nt) {
    int n = nt * 16 + col;
    float bv = bias[n];
    #pragma unroll
    for (int r = 0; r < 8; ++r) {
      int l = l0 + rbase + r;
      outp[(size_t)(bh * L_ + l) * HD_ + n] = (_Float16)(acc[nt][r] + bv);
    }
  }
}

// ---------------- V transpose: qkv f16 [M,3C] v-slice -> vT f16 [B,H,HD,L] ---
__global__ void vtrans_kernel(const _Float16* __restrict__ qkvh, _Float16* __restrict__ vT) {
  size_t idx = (size_t)blockIdx.x * 256 + threadIdx.x;  // B*H*HD*L elements
  int l = (int)(idx & (L_ - 1));
  size_t t = idx >> 11;
  int d  = (int)(t & (HD_ - 1));
  int bh = (int)(t >> 6);
  int b = bh >> 4, h = bh & 15;
  vT[idx] = qkvh[(size_t)(b * L_ + l) * C3_ + 2 * C_ + h * HD_ + d];
}

// ---------------- fused flash attention with frequency bias ------------------
// grid: (L/64, B*H), block 128 (4 waves). Each wave owns 16 query rows,
// streams 64-key tiles, online softmax, never spills logits to HBM.
__global__ void attn_kernel(const _Float16* __restrict__ qkvh, const _Float16* __restrict__ vT,
                            const _Float16* __restrict__ qb, const _Float16* __restrict__ kb,
                            const float* __restrict__ fscale, _Float16* __restrict__ o) {
  __shared__ _Float16 plds[4 * 16 * 64];  // per-wave 16x64 P tile for C->A relayout
  int bh = blockIdx.y;
  int b = bh >> 4, h = bh & 15;
  int wave = threadIdx.x >> 5, lane = threadIdx.x & 31;
  int col = lane & 15;
  int half  = (lane < 16) ? 0 : 8;
  int rbase = (lane < 16) ? 0 : 8;
  int l0 = blockIdx.x * 64 + wave * 16;
  float fs = fscale[0];

  // Preload Q and Qb fragments (HD=64 -> two K=32 chunks).
  int qrow = l0 + col;
  const _Float16* qbase  = qkvh + (size_t)(b * L_ + qrow) * C3_ + h * HD_;
  const _Float16* qbbase = qb   + (size_t)(bh * L_ + qrow) * HD_;
  v16h qfrag[2], qbfrag[2];
  #pragma unroll
  for (int kk = 0; kk < 2; ++kk) {
    qfrag[kk]  = load_frag16(qbase  + kk * 32, half);
    qbfrag[kk] = load_frag16(qbbase + kk * 32, half);
  }

  v8f oacc[4] = {};
  float m_run[8], l_run[8];
  #pragma unroll
  for (int r = 0; r < 8; ++r) { m_run[r] = -3.0e38f; l_run[r] = 0.f; }

  _Float16* myp = plds + wave * (16 * 64);

  for (int kt = 0; kt < L_ / 64; ++kt) {
    // ---- S = q@k^T and Sb = qb@kb^T over this 64-key tile ----
    v8f s[4] = {}, sb[4] = {};
    #pragma unroll
    for (int nt = 0; nt < 4; ++nt) {
      int key = kt * 64 + nt * 16 + col;
      const _Float16* kbase  = qkvh + (size_t)(b * L_ + key) * C3_ + C_ + h * HD_;
      const _Float16* kbbase = kb   + (size_t)(bh * L_ + key) * HD_;
      #pragma unroll
      for (int kk = 0; kk < 2; ++kk) {
        v16h kf  = load_frag16(kbase  + kk * 32, half);
        v16h kbf = load_frag16(kbbase + kk * 32, half);
        s[nt]  = __builtin_amdgcn_wmma_f32_16x16x32_f16(false, qfrag[kk],  false, kf,
                                                        (short)0, s[nt],  false, false);
        sb[nt] = __builtin_amdgcn_wmma_f32_16x16x32_f16(false, qbfrag[kk], false, kbf,
                                                        (short)0, sb[nt], false, false);
      }
    }
    // logits = S*scale + fs*Sb
    #pragma unroll
    for (int nt = 0; nt < 4; ++nt)
      #pragma unroll
      for (int r = 0; r < 8; ++r)
        s[nt][r] = s[nt][r] * SCALE_ + fs * sb[nt][r];

    // ---- online softmax: row stats via 16-lane shuffles ----
    float m_new[8], alpha[8], psum[8];
    #pragma unroll
    for (int r = 0; r < 8; ++r) {
      float mx = s[0][r];
      mx = fmaxf(mx, s[1][r]); mx = fmaxf(mx, s[2][r]); mx = fmaxf(mx, s[3][r]);
      #pragma unroll
      for (int off = 8; off >= 1; off >>= 1)
        mx = fmaxf(mx, __shfl_xor(mx, off, 16));
      m_new[r] = fmaxf(m_run[r], mx);
      alpha[r] = __expf(m_run[r] - m_new[r]);
      psum[r] = 0.f;
    }
    #pragma unroll
    for (int nt = 0; nt < 4; ++nt)
      #pragma unroll
      for (int r = 0; r < 8; ++r) {
        float p = __expf(s[nt][r] - m_new[r]);
        s[nt][r] = p;
        psum[r] += p;
      }
    #pragma unroll
    for (int r = 0; r < 8; ++r) {
      float ps = psum[r];
      #pragma unroll
      for (int off = 8; off >= 1; off >>= 1)
        ps += __shfl_xor(ps, off, 16);
      l_run[r] = l_run[r] * alpha[r] + ps;
      m_run[r] = m_new[r];
    }
    // rescale running output
    #pragma unroll
    for (int dt = 0; dt < 4; ++dt)
      #pragma unroll
      for (int r = 0; r < 8; ++r)
        oacc[dt][r] *= alpha[r];

    // ---- C-layout -> A-layout relayout of P via LDS ----
    __syncthreads();  // prior iteration's reads finished
    #pragma unroll
    for (int nt = 0; nt < 4; ++nt)
      #pragma unroll
      for (int r = 0; r < 8; ++r)
        myp[(rbase + r) * 64 + nt * 16 + col] = (_Float16)s[nt][r];
    __syncthreads();

    // ---- O += P @ V (V pre-transposed: contiguous keys per d column) ----
    #pragma unroll
    for (int kk = 0; kk < 2; ++kk) {
      v16h pf = load_frag16(myp + col * 64 + kk * 32, half);
      #pragma unroll
      for (int dt = 0; dt < 4; ++dt) {
        int d = dt * 16 + col;
        const _Float16* vb = vT + (size_t)(bh * HD_ + d) * L_ + kt * 64 + kk * 32;
        v16h vf = load_frag16(vb, half);
        oacc[dt] = __builtin_amdgcn_wmma_f32_16x16x32_f16(false, pf, false, vf,
                                                          (short)0, oacc[dt], false, false);
      }
    }
  }

  // ---- normalize and write O (f16, [M,C] layout, head slot h) ----
  #pragma unroll
  for (int dt = 0; dt < 4; ++dt)
    #pragma unroll
    for (int r = 0; r < 8; ++r) {
      int l = l0 + rbase + r;
      float v = oacc[dt][r] / l_run[r];
      o[(size_t)(b * L_ + l) * C_ + h * HD_ + dt * 16 + col] = (_Float16)v;
    }
}

// =============================== launcher ====================================
extern "C" void kernel_launch(void* const* d_in, const int* in_sizes, int n_in,
                              void* d_out, int out_size, void* d_ws, size_t ws_size,
                              hipStream_t stream) {
  const float* x        = (const float*)d_in[0];
  const float* freqd    = (const float*)d_in[1];
  const float* qkv_w    = (const float*)d_in[2];
  const float* qkv_b    = (const float*)d_in[3];
  const float* fp_w1    = (const float*)d_in[4];
  const float* fp_b1    = (const float*)d_in[5];
  const float* fp_ln_g  = (const float*)d_in[6];
  const float* fp_ln_b  = (const float*)d_in[7];
  const float* fp_w2    = (const float*)d_in[8];
  const float* fp_b2    = (const float*)d_in[9];
  const float* wq_w     = (const float*)d_in[10];
  const float* wq_b     = (const float*)d_in[11];
  const float* wk_w     = (const float*)d_in[12];
  const float* wk_b     = (const float*)d_in[13];
  const float* out_w    = (const float*)d_in[14];
  const float* out_b    = (const float*)d_in[15];
  const float* n1_g     = (const float*)d_in[16];
  const float* n1_b     = (const float*)d_in[17];
  const float* n2_g     = (const float*)d_in[18];
  const float* n2_b     = (const float*)d_in[19];
  const float* mlp_w1   = (const float*)d_in[20];
  const float* mlp_b1   = (const float*)d_in[21];
  const float* mlp_w2   = (const float*)d_in[22];
  const float* mlp_b2   = (const float*)d_in[23];
  const float* fscale   = (const float*)d_in[24];

  // ---- workspace carve-up (all f16 unless noted) ----
  char* base = (char*)d_ws;
  size_t off = 0;
  auto alloc = [&](size_t bytes) -> char* {
    char* p = base + off;
    off = (off + bytes + 255) & ~(size_t)255;
    return p;
  };
  _Float16* qkvT   = (_Float16*)alloc((size_t)C3_ * C_  * 2);  // [3C, C]
  _Float16* fpw2T  = (_Float16*)alloc((size_t)C_  * C_  * 2);  // [C, C]
  _Float16* wqT    = (_Float16*)alloc((size_t)HD_ * HD_ * 2);
  _Float16* wkT    = (_Float16*)alloc((size_t)HD_ * HD_ * 2);
  _Float16* outT   = (_Float16*)alloc((size_t)C_  * C_  * 2);
  _Float16* mlp1T  = (_Float16*)alloc((size_t)FF_ * C_  * 2);  // [FF, C]
  _Float16* mlp2T  = (_Float16*)alloc((size_t)C_  * FF_ * 2);  // [C, FF]
  _Float16* xn     = (_Float16*)alloc((size_t)M_  * C_  * 2);
  _Float16* qkvh   = (_Float16*)alloc((size_t)M_  * C3_ * 2);
  _Float16* fbact  = (_Float16*)alloc((size_t)M_  * C_  * 2);
  _Float16* fb2    = (_Float16*)alloc((size_t)M_  * C_  * 2);
  _Float16* qbbuf  = (_Float16*)alloc((size_t)M_  * C_  * 2);  // [B,H,L,HD]
  _Float16* kbbuf  = (_Float16*)alloc((size_t)M_  * C_  * 2);  // [B,H,L,HD]
  _Float16* vTbuf  = (_Float16*)alloc((size_t)M_  * C_  * 2);  // [B,H,HD,L]
  _Float16* o16    = (_Float16*)alloc((size_t)M_  * C_  * 2);
  float*    x1     = (float*)   alloc((size_t)M_  * C_  * 4);
  _Float16* xn2    = (_Float16*)alloc((size_t)M_  * C_  * 2);
  _Float16* hbuf   = (_Float16*)alloc((size_t)M_  * FF_ * 2);
  (void)ws_size; (void)in_sizes; (void)n_in; (void)out_size;

  auto nblk = [](size_t n) { return (unsigned)((n + 255) / 256); };

  // 1) weight transposes to f16 (stay L2-resident across the GEMMs)
  wtrans_kernel<<<nblk((size_t)C_ * C3_), 256, 0, stream>>>(qkv_w,  qkvT,  C_,  C3_);
  wtrans_kernel<<<nblk((size_t)C_ * C_ ), 256, 0, stream>>>(fp_w2,  fpw2T, C_,  C_);
  wtrans_kernel<<<nblk((size_t)HD_* HD_), 256, 0, stream>>>(wq_w,   wqT,   HD_, HD_);
  wtrans_kernel<<<nblk((size_t)HD_* HD_), 256, 0, stream>>>(wk_w,   wkT,   HD_, HD_);
  wtrans_kernel<<<nblk((size_t)C_ * C_ ), 256, 0, stream>>>(out_w,  outT,  C_,  C_);
  wtrans_kernel<<<nblk((size_t)C_ * FF_), 256, 0, stream>>>(mlp_w1, mlp1T, C_,  FF_);
  wtrans_kernel<<<nblk((size_t)FF_* C_ ), 256, 0, stream>>>(mlp_w2, mlp2T, FF_, C_);

  // 2) LN1 -> xn (f16)
  ln_kernel<<<M_, 256, 0, stream>>>(x, n1_g, n1_b, xn);

  // 3) QKV projection (f16 out)
  gemm_kernel<0,0,1><<<dim3(C3_/128, M_/128), 256, 0, stream>>>(
      xn, qkvT, qkv_b, nullptr, qkvh, M_, C3_, C_);

  // 4) frequency-bias front: rank-1 + LN + GeLU fused
  freqfb_kernel<<<M_, 256, 0, stream>>>(freqd, fp_w1, fp_b1, fp_ln_g, fp_ln_b, fbact);

  // 5) fb = act @ fp_w2 + fp_b2
  gemm_kernel<0,0,1><<<dim3(C_/128, M_/128), 256, 0, stream>>>(
      fbact, fpw2T, fp_b2, nullptr, fb2, M_, C_, C_);

  // 6) qb / kb per-head 64x64 projections
  headproj_kernel<<<dim3(L_/64, B_*H_), 128, 0, stream>>>(fb2, wqT, wq_b, qbbuf);
  headproj_kernel<<<dim3(L_/64, B_*H_), 128, 0, stream>>>(fb2, wkT, wk_b, kbbuf);

  // 7) V transpose for contiguous P@V B-fragments
  vtrans_kernel<<<nblk((size_t)M_ * C_), 256, 0, stream>>>(qkvh, vTbuf);

  // 8) fused flash attention with frequency bias
  attn_kernel<<<dim3(L_/64, B_*H_), 128, 0, stream>>>(
      qkvh, vTbuf, qbbuf, kbbuf, fscale, o16);

  // 9) output projection + residual (f32)
  gemm_kernel<0,1,0><<<dim3(C_/128, M_/128), 256, 0, stream>>>(
      o16, outT, out_b, x, x1, M_, C_, C_);

  // 10) LN2 -> xn2 (f16)
  ln_kernel<<<M_, 256, 0, stream>>>(x1, n2_g, n2_b, xn2);

  // 11) MLP up + exact GeLU (f16 out)
  gemm_kernel<1,0,1><<<dim3(FF_/128, M_/128), 256, 0, stream>>>(
      xn2, mlp1T, mlp_b1, nullptr, hbuf, M_, FF_, C_);

  // 12) MLP down + residual -> d_out (f32)
  gemm_kernel<0,1,0><<<dim3(C_/128, M_/128), 256, 0, stream>>>(
      hbuf, mlp2T, mlp_b2, x1, (float*)d_out, M_, C_, FF_);
}
